// CodeBook_36833639530907
// MI455X (gfx1250) — matrix operand order repaired and checked by reading, not compiled, not run on hardware
//
#include <hip/hip_runtime.h>

#define THREADS        128          // 4 waves (wave32)
#define TOKENS_PER_WG  64           // 16 tokens per wave
#define C_DIM          256
#define K_CODES        1024
#define LDS_STRIDE     264          // halfs per z row (pad: 16B align + bank spread)
#define EB_STRIDE      264          // halfs per staged code row
#define N_SPATIAL      32768L       // 32*32*32
#define TOTAL_ELEMS    33554432.0f  // 4*256*32768

typedef __attribute__((ext_vector_type(16))) __bf16 v16bf;
typedef __attribute__((ext_vector_type(8)))  float  v8f;
typedef __attribute__((ext_vector_type(4)))  int    v4i;

struct alignas(16) U4 { unsigned x, y, z, w; };
union ABfrag { v16bf v; U4 q[2]; };

#if defined(__HIP_DEVICE_COMPILE__) && __has_builtin(__builtin_amdgcn_global_load_async_to_lds_b128)
#define HAS_ASYNC_LDS 1
typedef __attribute__((address_space(1))) v4i* gv4i_p;   // global int4*
typedef __attribute__((address_space(3))) v4i* lv4i_p;   // LDS int4*
#else
#define HAS_ASYNC_LDS 0
#endif

__device__ __forceinline__ void wait_async_all() {
#if HAS_ASYNC_LDS
#if __has_builtin(__builtin_amdgcn_s_wait_asynccnt)
    __builtin_amdgcn_s_wait_asynccnt(0);
#else
    asm volatile("s_wait_asynccnt 0x0" ::: "memory");
#endif
#endif
}

__device__ __forceinline__ unsigned short f32_to_bf16_rne(float f) {
    unsigned u = __float_as_uint(f);
    unsigned r = u + 0x7FFFu + ((u >> 16) & 1u);
    return (unsigned short)(r >> 16);
}

// Stage one 16-code x 256-ch bf16 tile (8 KB) into LDS; async when available.
__device__ __forceinline__ void stage_btile(const unsigned short* __restrict__ ebuf,
                                            unsigned short* dst, int ct, int tid) {
    // 16 codes x 32 16B-chunks = 512 chunks; 4 per thread
#pragma unroll
    for (int r = 0; r < 4; ++r) {
        int chunk = tid + r * THREADS;          // 0..511
        int codeL = chunk >> 5;
        int off   = (chunk & 31) * 8;           // halfs
        const unsigned short* src = ebuf + (size_t)(ct * 16 + codeL) * C_DIM + off;
        unsigned short*       d   = dst + codeL * EB_STRIDE + off;
#if HAS_ASYNC_LDS
        __builtin_amdgcn_global_load_async_to_lds_b128(
            (gv4i_p)src, (lv4i_p)d, 0, 0);
#else
        *(U4*)d = *(const U4*)src;
#endif
    }
}

// ---- Kernel 1: codebook fp32 -> bf16, -0.5*||e||^2, zero loss accumulator ----
__global__ void vq_prep(const float* __restrict__ emb,
                        unsigned short* __restrict__ ebuf,
                        float* __restrict__ nhesq,
                        float* __restrict__ lossAcc) {
    int k = blockIdx.x;
    int c = threadIdx.x;
    float v = emb[k * C_DIM + c];
    ebuf[k * C_DIM + c] = f32_to_bf16_rne(v);
    __shared__ float red[C_DIM];
    red[c] = v * v;
    __syncthreads();
    for (int s = C_DIM / 2; s > 0; s >>= 1) {
        if (c < s) red[c] += red[c + s];
        __syncthreads();
    }
    if (c == 0) nhesq[k] = -0.5f * red[0];
    if (k == 0 && c == 0) *lossAcc = 0.0f;
}

// ---- Kernel 2: WMMA argmax of (z.e - 0.5||e||^2), gather, loss partials ----
__global__ __launch_bounds__(THREADS)
void vq_main(const float* __restrict__ ze,
             const float* __restrict__ emb,
             const unsigned short* __restrict__ ebuf,
             const float* __restrict__ nhesq,
             float* __restrict__ out,
             float* __restrict__ lossAcc) {
    __shared__ unsigned short zt[TOKENS_PER_WG * LDS_STRIDE];   // 33.8 KB
    __shared__ unsigned short ebt[2][16 * EB_STRIDE];           // 2 x 8.25 KB
    __shared__ int   codes[TOKENS_PER_WG];
    __shared__ float lred[THREADS];

    const int  tid  = threadIdx.x;
    const long tok0 = (long)blockIdx.x * TOKENS_PER_WG;
    const long b    = tok0 >> 15;
    const long n0   = tok0 & (N_SPATIAL - 1);
    const float* zbase = ze + b * (C_DIM * N_SPATIAL) + n0;

    // Stage z tile: fp32 [c][n] (coalesced) -> LDS bf16 token-major [t][c]
    for (int i = tid; i < C_DIM * TOKENS_PER_WG; i += THREADS) {
        int c = i / TOKENS_PER_WG;
        int t = i % TOKENS_PER_WG;
        float v = zbase[(long)c * N_SPATIAL + t];
        zt[t * LDS_STRIDE + c] = f32_to_bf16_rne(v);
    }
    // Prefill B buffer 0 (overlaps with z staging above).
    stage_btile(ebuf, ebt[0], 0, tid);
    wait_async_all();
    __syncthreads();

    const int lane = tid & 31;
    const int wave = tid >> 5;
    const int mrow = lane & 15;     // A: token row; B/C/D: code column
    const int hi   = lane >> 4;
    const int tloc = wave * 16 + mrow;

    // Hoist all A fragments (16 tokens x 256 ch = 8 chunks of K=32) into VGPRs.
    ABfrag a[8];
    {
        const unsigned short* rowp = &zt[tloc * LDS_STRIDE];
        const int kb = hi * 16;
#pragma unroll
        for (int cc = 0; cc < 8; ++cc) {
            a[cc].q[0] = *(const U4*)(rowp + cc * 32 + kb);
            a[cc].q[1] = *(const U4*)(rowp + cc * 32 + kb + 8);
        }
    }

    float bestv[8];
    int   besti[8];
#pragma unroll
    for (int i = 0; i < 8; ++i) { bestv[i] = -3.4e38f; besti[i] = 0; }

    float seed_carry = nhesq[mrow];                 // seed for ct = 0

    for (int ct = 0; ct < K_CODES / 16; ++ct) {
        const unsigned short* cur = ebt[ct & 1];
        // Prefetch next code tile (async fill overlaps the WMMA chain below).
        if (ct + 1 < K_CODES / 16)
            stage_btile(ebuf, ebt[(ct + 1) & 1], ct + 1, tid);

        const float seed = seed_carry;
        if (ct + 1 < K_CODES / 16)
            seed_carry = nhesq[(ct + 1) * 16 + mrow];

        // Two independent accumulator chains for matrix-pipe ILP.
        v8f acc0, acc1;
#pragma unroll
        for (int i = 0; i < 8; ++i) { acc0[i] = seed; acc1[i] = 0.0f; }

        const unsigned short* brow = cur + mrow * EB_STRIDE + hi * 16;
#pragma unroll
        for (int cc = 0; cc < 4; ++cc) {
            ABfrag bf;
            bf.q[0] = *(const U4*)(brow + cc * 32);
            bf.q[1] = *(const U4*)(brow + cc * 32 + 8);
            acc0 = __builtin_amdgcn_wmma_f32_16x16x32_bf16(
                false, a[cc].v, false, bf.v, (short)0, acc0, false, false);
        }
#pragma unroll
        for (int cc = 4; cc < 8; ++cc) {
            ABfrag bf;
            bf.q[0] = *(const U4*)(brow + cc * 32);
            bf.q[1] = *(const U4*)(brow + cc * 32 + 8);
            acc1 = __builtin_amdgcn_wmma_f32_16x16x32_bf16(
                false, a[cc].v, false, bf.v, (short)0, acc1, false, false);
        }

        const int code = ct * 16 + mrow;
#pragma unroll
        for (int i = 0; i < 8; ++i) {
            float sc = acc0[i] + acc1[i];
            if (sc > bestv[i]) { bestv[i] = sc; besti[i] = code; }
        }

        // Next tile fill must be complete, and all waves done reading `cur`
        // before it is refilled at iteration ct+1.
        wait_async_all();
        __syncthreads();
    }

    // Reduce across the 16 lanes holding the same 8 token rows.
#pragma unroll
    for (int d = 1; d < 16; d <<= 1) {
#pragma unroll
        for (int i = 0; i < 8; ++i) {
            float ov = __shfl_xor(bestv[i], d, 32);
            int   oi = __shfl_xor(besti[i], d, 32);
            if (ov > bestv[i] || (ov == bestv[i] && oi < besti[i])) {
                bestv[i] = ov; besti[i] = oi;
            }
        }
    }
    if (mrow == 0) {
#pragma unroll
        for (int i = 0; i < 8; ++i)
            codes[wave * 16 + hi * 8 + i] = besti[i];
    }
    __syncthreads();

    // Output: z_q_st == emb[code] exactly (fp32 gather); loss partials.
    float lsum = 0.0f;
    for (int i = tid; i < C_DIM * TOKENS_PER_WG; i += THREADS) {
        int c = i / TOKENS_PER_WG;
        int t = i % TOKENS_PER_WG;
        int code = codes[t];
        float qv = emb[code * C_DIM + c];
        long  go = (long)c * N_SPATIAL + t;
        float zv = zbase[go];
        out[b * (C_DIM * N_SPATIAL) + n0 + go] = qv;
        float d = zv - qv;
        lsum += d * d;
    }
    lred[tid] = lsum;
    __syncthreads();
    for (int s = THREADS / 2; s > 0; s >>= 1) {
        if (tid < s) lred[tid] += lred[tid + s];
        __syncthreads();
    }
    if (tid == 0) atomicAdd(lossAcc, lred[0]);
}

// ---- Kernel 3: finalize loss = (1 + BETA) * mean(sq err) ----
__global__ void vq_fin(const float* __restrict__ lossAcc,
                       float* __restrict__ out_loss) {
    *out_loss = 1.25f * (*lossAcc) / TOTAL_ELEMS;
}

extern "C" void kernel_launch(void* const* d_in, const int* in_sizes, int n_in,
                              void* d_out, int out_size, void* d_ws, size_t ws_size,
                              hipStream_t stream) {
    const float* ze  = (const float*)d_in[0];   // (4,256,32,32,32) fp32
    const float* emb = (const float*)d_in[1];   // (1024,256) fp32
    float* out = (float*)d_out;                 // 33554432 z_q_st + 1 loss

    char* ws = (char*)d_ws;
    unsigned short* ebuf  = (unsigned short*)ws;                 // 512 KB
    float*          nhesq = (float*)(ws + (size_t)K_CODES * C_DIM * 2);
    float*          lacc  = (float*)(ws + (size_t)K_CODES * C_DIM * 2
                                        + (size_t)K_CODES * 4);

    vq_prep<<<K_CODES, C_DIM, 0, stream>>>(emb, ebuf, nhesq, lacc);

    const int n_tiles = (4 * (int)N_SPATIAL) / TOKENS_PER_WG;    // 2048
    vq_main<<<n_tiles, THREADS, 0, stream>>>(ze, emb, ebuf, nhesq, out, lacc);

    vq_fin<<<1, 1, 0, stream>>>(lacc, out + (out_size - 1));
}
